// GraphSAGEModel_36197984370745
// MI455X (gfx1250) — matrix operand order repaired and checked by reading, not compiled, not run on hardware
//
#include <hip/hip_runtime.h>
#include <hip/hip_bf16.h>

// ---------------------------------------------------------------------------
// GraphSAGE forward for MI455X (gfx1250, wave32, WMMA 16x16x32 f16 -> f32 acc)
// ---------------------------------------------------------------------------

typedef __attribute__((ext_vector_type(16))) _Float16 v16h;
typedef __attribute__((ext_vector_type(8)))  _Float16 v8h;
typedef __attribute__((ext_vector_type(8)))  float    v8f;

#define N_NODES 50000
#define N_EDGES 800000

static __device__ __forceinline__ v8f wmma16(v16h a, v16h b, v8f c) {
  // D = A(16x32 f16) * B(32x16 f16) + C(16x16 f32)
  return __builtin_amdgcn_wmma_f32_16x16x32_f16(
      /*neg_a=*/false, a, /*neg_b=*/false, b,
      /*c_mod=*/(short)0, c, /*reuse_a=*/false, /*reuse_b=*/false);
}

static __device__ __forceinline__ v8f vzero8() {
  v8f z;
#pragma unroll
  for (int i = 0; i < 8; ++i) z[i] = 0.0f;
  return z;
}

// A fragment (16x32 f16). ISA layout: lanes 0-15 = M rows with K {0..7,16..23},
// lanes 16-31 = same M rows with K {8..15,24..31}. row = Ah + (lane&15)*stride,
// kb = ktile*32 + (lane>>4)*8.
static __device__ __forceinline__ v16h load_a_frag(const _Float16* row, int kb) {
  v8h lo = *(const v8h*)(row + kb);
  v8h hh = *(const v8h*)(row + kb + 16);
  v16h a;
#pragma unroll
  for (int i = 0; i < 8; ++i) { a[i] = lo[i]; a[i + 8] = hh[i]; }
  return a;
}

// B fragment (32x16 f16). Lane l holds column N = l&15, K {hi8..hi8+7, 16+hi8..16+hi8+7}.
// p = W + ktile*32*ncols + ntile*16 + (lane&15)   (row-major [K][N] f16 weights)
static __device__ __forceinline__ v16h load_b_frag(const _Float16* p, int ld, int hi8) {
  v16h b;
#pragma unroll
  for (int i = 0; i < 8; ++i) { b[i] = p[(hi8 + i) * ld]; b[i + 8] = p[(16 + hi8 + i) * ld]; }
  return b;
}

// ------------------------------- prep kernels ------------------------------

__global__ void k_zero_f32(float* p, int n) {
  int i = blockIdx.x * blockDim.x + threadIdx.x;
  if (i < n) p[i] = 0.0f;
}

__global__ void k_cvt_f16(const float* __restrict__ in, _Float16* __restrict__ out, int n) {
  int i = blockIdx.x * blockDim.x + threadIdx.x;
  if (i < n) out[i] = (_Float16)in[i];
}

// Pack conv weights vertically: rows 0..63 = Wl, rows 64..127 = Wr -> [128][64] f16
__global__ void k_pack_conv_w(const float* __restrict__ Wl, const float* __restrict__ Wr,
                              _Float16* __restrict__ out) {
  int i = blockIdx.x * blockDim.x + threadIdx.x;
  if (i >= 128 * 64) return;
  int r = i >> 6, c = i & 63;
  float v = (r < 64) ? Wl[r * 64 + c] : Wr[(r - 64) * 64 + c];
  out[i] = (_Float16)v;
}

// Pad fcout_W [192,1] -> [192,16] f16 (col 0 = weights, rest 0)
__global__ void k_pad_wout(const float* __restrict__ W, _Float16* __restrict__ out) {
  int i = blockIdx.x * blockDim.x + threadIdx.x;
  if (i >= 192 * 16) return;
  int r = i >> 4, c = i & 15;
  out[i] = (_Float16)((c == 0) ? W[r] : 0.0f);
}

// ------------------------------ graph kernels ------------------------------

__global__ void k_degree(const int* __restrict__ ei, float* deg) {
  int e = blockIdx.x * blockDim.x + threadIdx.x;
  if (e < N_EDGES) atomicAdd(&deg[ei[N_EDGES + e]], 1.0f);
}

__global__ void k_invdeg(float* deg) {
  int i = blockIdx.x * blockDim.x + threadIdx.x;
  if (i < N_NODES) deg[i] = 1.0f / fmaxf(deg[i], 1.0f);
}

// aggr[dst] += x[src]   (4 floats per thread)
__global__ void k_scatter(const int* __restrict__ ei, const float* __restrict__ x,
                          float* aggr) {
  int t = blockIdx.x * blockDim.x + threadIdx.x;
  if (t >= N_EDGES * 16) return;
  int e = t >> 4, q = (t & 15) << 2;
  int s = ei[e], d = ei[N_EDGES + e];
  const float* xs = x + s * 64 + q;
  float* ap = aggr + d * 64 + q;
  atomicAdd(ap + 0, xs[0]);
  atomicAdd(ap + 1, xs[1]);
  atomicAdd(ap + 2, xs[2]);
  atomicAdd(ap + 3, xs[3]);
}

// --------------------------- SAGEConv GEMM kernel --------------------------
// x_new = [aggr*inv_deg | x] @ [Wl;Wr] + b   per 16-node tile, one wave/tile.
#define CSTRIDE 136  // 128 cols + 8 pad halfs, multiple of 8 (16B aligned rows)

__global__ void __launch_bounds__(128) k_conv_gemm(
    const float* __restrict__ aggr, const float* __restrict__ invdeg,
    const float* __restrict__ xin, const _Float16* __restrict__ Wc,
    const float* __restrict__ bias, float* __restrict__ outF,
    _Float16* __restrict__ outH) {
  __shared__ __align__(16) _Float16 smem[4 * 16 * CSTRIDE];
  const int lane = threadIdx.x & 31;
  const int wave = threadIdx.x >> 5;
  const int base = (blockIdx.x * 4 + wave) * 16;
  if (base >= N_NODES) return;  // uniform per wave -> EXEC stays all-ones
  _Float16* Ah = smem + wave * 16 * CSTRIDE;

  // Stage A rows: cols 0..63 = aggr*inv_deg, cols 64..127 = x  (f32 -> f16)
  for (int i = lane; i < 256; i += 32) {
    int m = i >> 4, c = i & 15;
    int node = base + m;
    const float* sp = ((c < 8) ? aggr : xin) + node * 64 + (c & 7) * 8;
    float s = (c < 8) ? invdeg[node] : 1.0f;
    v8h h;
#pragma unroll
    for (int j = 0; j < 8; ++j) h[j] = (_Float16)(sp[j] * s);
    *(v8h*)(Ah + m * CSTRIDE + c * 8) = h;
  }
  asm volatile("s_wait_dscnt 0" ::: "memory");  // in-wave LDS fence

  const int nl = lane & 15;
  const int hi8 = (lane >> 4) * 8;
  const _Float16* arow = Ah + nl * CSTRIDE;  // A row M = lane&15

  v8f acc[4];
#pragma unroll
  for (int nt = 0; nt < 4; ++nt) acc[nt] = vzero8();
#pragma unroll
  for (int kt = 0; kt < 4; ++kt) {
    v16h a = load_a_frag(arow, kt * 32 + hi8);
    const _Float16* wp = Wc + (kt * 32) * 64 + nl;
#pragma unroll
    for (int nt = 0; nt < 4; ++nt)
      acc[nt] = wmma16(a, load_b_frag(wp + nt * 16, 64, hi8), acc[nt]);
  }
  // C layout: VGPR r -> (M = hi8 + r, N = nt*16 + nl). LeakyReLU(1) == identity.
#pragma unroll
  for (int nt = 0; nt < 4; ++nt) {
    int n = nt * 16 + nl;
    float bb = bias[n];
#pragma unroll
    for (int r = 0; r < 8; ++r) {
      int node = base + hi8 + r;
      float v = acc[nt][r] + bb;
      if (outF) outF[node * 64 + n] = v;
      if (outH) outH[node * 64 + n] = (_Float16)v;
    }
  }
}

// ------------------------------ edge MLP kernel ----------------------------
// Per 16-edge tile (one wave): A = [x2[src]|x2[dst]|x[src]|x[dst]] (16x256 f16)
// fc0: A[16,256]@W0[256,128] -> overwrite cols 0..127
// fc1: A[16,256]@W1[256,64]  -> overwrite cols 0..63
// fcout: K map (k<64 -> cols 0..63, k>=64 -> cols 128..255), Wout padded [192,16]
#define ESTRIDE 264  // 256 + 8 pad halfs

__global__ void __launch_bounds__(128) k_edge_mlp(
    const int* __restrict__ ei, const _Float16* __restrict__ x0h,
    const _Float16* __restrict__ x2h, const _Float16* __restrict__ W0,
    const float* __restrict__ b0, const _Float16* __restrict__ W1,
    const float* __restrict__ b1, const _Float16* __restrict__ Wo,
    const float* __restrict__ bo, float* __restrict__ out) {
  __shared__ __align__(16) _Float16 smem[4 * 16 * ESTRIDE];
  const int lane = threadIdx.x & 31;
  const int wave = threadIdx.x >> 5;
  const int ebase = (blockIdx.x * 4 + wave) * 16;  // E % 64 == 0: no tail
  _Float16* Ah = smem + wave * 16 * ESTRIDE;

  // Stage: 16 rows x 16 chunks of 16 halfs (gathers hit L2: x tables ~12.8 MB)
  for (int i = lane; i < 256; i += 32) {
    int m = i >> 4, c = i & 15;
    int e = ebase + m;
    int node = ((c >> 2) & 1) ? ei[N_EDGES + e] : ei[e];  // chunks 0-3 src,4-7 dst,...
    const _Float16* sp = ((c < 8) ? x2h : x0h) + node * 64 + (c & 3) * 16;
    v8h h0 = *(const v8h*)sp;
    v8h h1 = *(const v8h*)(sp + 8);
    _Float16* dp = Ah + m * ESTRIDE + c * 16;
    *(v8h*)dp = h0;
    *(v8h*)(dp + 8) = h1;
  }
  asm volatile("s_wait_dscnt 0" ::: "memory");

  const int nl = lane & 15;
  const int hi8 = (lane >> 4) * 8;
  const _Float16* arow = Ah + nl * ESTRIDE;

  // ---- fc0: [16,256] @ [256,128] ----
  v8f acc0[8];
#pragma unroll
  for (int nt = 0; nt < 8; ++nt) acc0[nt] = vzero8();
#pragma unroll
  for (int kt = 0; kt < 8; ++kt) {
    v16h a = load_a_frag(arow, kt * 32 + hi8);
    const _Float16* wp = W0 + (kt * 32) * 128 + nl;
#pragma unroll
    for (int nt = 0; nt < 8; ++nt)
      acc0[nt] = wmma16(a, load_b_frag(wp + nt * 16, 128, hi8), acc0[nt]);
  }
#pragma unroll
  for (int nt = 0; nt < 8; ++nt) {
    float bb = b0[nt * 16 + nl];
#pragma unroll
    for (int r = 0; r < 8; ++r)
      Ah[(hi8 + r) * ESTRIDE + nt * 16 + nl] = (_Float16)(acc0[nt][r] + bb);
  }
  asm volatile("s_wait_dscnt 0" ::: "memory");

  // ---- fc1: [16,256] @ [256,64] ----
  v8f acc1[4];
#pragma unroll
  for (int nt = 0; nt < 4; ++nt) acc1[nt] = vzero8();
#pragma unroll
  for (int kt = 0; kt < 8; ++kt) {
    v16h a = load_a_frag(arow, kt * 32 + hi8);
    const _Float16* wp = W1 + (kt * 32) * 64 + nl;
#pragma unroll
    for (int nt = 0; nt < 4; ++nt)
      acc1[nt] = wmma16(a, load_b_frag(wp + nt * 16, 64, hi8), acc1[nt]);
  }
#pragma unroll
  for (int nt = 0; nt < 4; ++nt) {
    float bb = b1[nt * 16 + nl];
#pragma unroll
    for (int r = 0; r < 8; ++r)
      Ah[(hi8 + r) * ESTRIDE + nt * 16 + nl] = (_Float16)(acc1[nt][r] + bb);
  }
  asm volatile("s_wait_dscnt 0" ::: "memory");

  // ---- fcout: [16,192] @ [192,16(pad)] ----
  v8f acco = vzero8();
#pragma unroll
  for (int kt = 0; kt < 6; ++kt) {
    int kcol = (kt < 2) ? kt * 32 : kt * 32 + 64;  // logical k -> physical col
    v16h a = load_a_frag(arow, kcol + hi8);
    acco = wmma16(a, load_b_frag(Wo + (kt * 32) * 16 + nl, 16, hi8), acco);
  }
  // N=0 column lives in lanes 0 (M=0..7) and 16 (M=8..15)
  if (nl == 0) {
    float bb = bo[0];
#pragma unroll
    for (int r = 0; r < 8; ++r) out[ebase + hi8 + r] = acco[r] + bb;
  }
}

// ------------------------------- launcher ----------------------------------

extern "C" void kernel_launch(void* const* d_in, const int* in_sizes, int n_in,
                              void* d_out, int out_size, void* d_ws, size_t ws_size,
                              hipStream_t stream) {
  (void)in_sizes; (void)n_in; (void)out_size; (void)ws_size;
  const float* x    = (const float*)d_in[0];
  const int*   ei   = (const int*)d_in[1];
  const float* c0Wl = (const float*)d_in[2];
  const float* c0Wr = (const float*)d_in[3];
  const float* c0b  = (const float*)d_in[4];
  const float* c1Wl = (const float*)d_in[5];
  const float* c1Wr = (const float*)d_in[6];
  const float* c1b  = (const float*)d_in[7];
  const float* fc0W = (const float*)d_in[8];
  const float* fc0b = (const float*)d_in[9];
  const float* fc1W = (const float*)d_in[10];
  const float* fc1b = (const float*)d_in[11];
  const float* foW  = (const float*)d_in[12];
  const float* fob  = (const float*)d_in[13];
  float* out = (float*)d_out;

  char* ws = (char*)d_ws;
  size_t off = 0;
  auto take = [&](size_t bytes) {
    char* p = ws + off;
    off = (off + bytes + 255) & ~(size_t)255;
    return p;
  };
  float*    deg  = (float*)take((size_t)N_NODES * 4);
  float*    aggr = (float*)take((size_t)N_NODES * 64 * 4);
  float*    x1   = (float*)take((size_t)N_NODES * 64 * 4);
  _Float16* x0h  = (_Float16*)take((size_t)N_NODES * 64 * 2);
  _Float16* x2h  = (_Float16*)take((size_t)N_NODES * 64 * 2);
  _Float16* Wc0  = (_Float16*)take(128 * 64 * 2);
  _Float16* Wc1  = (_Float16*)take(128 * 64 * 2);
  _Float16* W0h  = (_Float16*)take(256 * 128 * 2);
  _Float16* W1h  = (_Float16*)take(256 * 64 * 2);
  _Float16* WoH  = (_Float16*)take(192 * 16 * 2);

  const int B = 256;
  // precision prep (f32 -> f16 activations & weights)
  k_cvt_f16<<<(N_NODES * 64 + B - 1) / B, B, 0, stream>>>(x, x0h, N_NODES * 64);
  k_pack_conv_w<<<(128 * 64 + B - 1) / B, B, 0, stream>>>(c0Wl, c0Wr, Wc0);
  k_pack_conv_w<<<(128 * 64 + B - 1) / B, B, 0, stream>>>(c1Wl, c1Wr, Wc1);
  k_cvt_f16<<<(256 * 128 + B - 1) / B, B, 0, stream>>>(fc0W, W0h, 256 * 128);
  k_cvt_f16<<<(256 * 64 + B - 1) / B, B, 0, stream>>>(fc1W, W1h, 256 * 64);
  k_pad_wout<<<(192 * 16 + B - 1) / B, B, 0, stream>>>(foW, WoH);

  // degrees + mean normalization
  k_zero_f32<<<(N_NODES + B - 1) / B, B, 0, stream>>>(deg, N_NODES);
  k_degree<<<(N_EDGES + B - 1) / B, B, 0, stream>>>(ei, deg);
  k_invdeg<<<(N_NODES + B - 1) / B, B, 0, stream>>>(deg);

  const int convBlocks = (N_NODES / 16 + 3) / 4;  // 3125 wave-tiles / 4 waves
  // conv0
  k_zero_f32<<<(N_NODES * 64 + B - 1) / B, B, 0, stream>>>(aggr, N_NODES * 64);
  k_scatter<<<(N_EDGES * 16 + B - 1) / B, B, 0, stream>>>(ei, x, aggr);
  k_conv_gemm<<<convBlocks, 128, 0, stream>>>(aggr, deg, x, Wc0, c0b, x1, nullptr);
  // conv1
  k_zero_f32<<<(N_NODES * 64 + B - 1) / B, B, 0, stream>>>(aggr, N_NODES * 64);
  k_scatter<<<(N_EDGES * 16 + B - 1) / B, B, 0, stream>>>(ei, x1, aggr);
  k_conv_gemm<<<convBlocks, 128, 0, stream>>>(aggr, deg, x1, Wc1, c1b, nullptr, x2h);

  // fused edge MLP (dominant ~74 GFLOP) — 102 WMMAs per 16-edge tile
  k_edge_mlp<<<N_EDGES / 64, 128, 0, stream>>>(ei, x0h, x2h, W0h, fc0b, W1h, fc1b,
                                               WoH, fob, out);
}
